// Attention_59837484368330
// MI455X (gfx1250) — matrix operand order repaired and checked by reading, not compiled, not run on hardware
//
#include <hip/hip_runtime.h>

typedef _Float16 f16;
typedef __attribute__((ext_vector_type(8)))  f16   h8;
typedef __attribute__((ext_vector_type(16))) f16   v16h;
typedef __attribute__((ext_vector_type(8)))  float v8f;

#define B_    256
#define NTOK  164
#define NS_   100
#define DIM_  256
#define H_    8
#define D_    32
#define HID_  50
#define NE_   4
#define MPAD  176     // padded query rows (11 tiles)
#define NPAD  192     // padded key cols  (12 tiles, K-chain of 6x32)
#define LDSD  196     // s_dots row stride (floats)
#define NEG_MAX (-3.402823466e38f)

// ---- workspace layout (bytes) ----
#define OFF_WQKS 0ul                    // 512x256 f16
#define OFF_WQKQ 262144ul               // 512x256 f16
#define OFF_WV   524288ul               // 256x256 f16
#define OFF_WOUT 655360ul               // 256x256 f16
#define OFF_W1T  786432ul               // NE x 64 x 128 f16 (w1^T, padded)
#define OFF_W2T  851968ul               // NE x 112 x 64 f16 (w2^T, padded)
#define OFF_QH   909312ul               // (B,H,164,32) f16
#define OFF_KH   22405120ul             // (B,H,164,32) f16
#define OFF_VT   43900928ul             // (B,H,32,176) f16  (V transposed)
#define OFF_AO   66969600ul             // (B,164,256) f16  (attn out pre-proj)
// total ~88.5 MB

union F16x16 { v16h v; h8 h[2]; };

// A-operand (16x32 f16, row-major source): lane holds row (lane&15),
// K chunks {kb..kb+7} and {kb+16..kb+23}, kb = koff + (lane>>4)*8.
__device__ inline v16h ldA(const f16* p, int ld, int mbase, int koff) {
  const int lane = threadIdx.x & 31;
  const f16* q = p + (size_t)(mbase + (lane & 15)) * ld + koff + ((lane >> 4) << 3);
  F16x16 u;
  u.h[0] = *(const h8*)(q);
  u.h[1] = *(const h8*)(q + 16);
  return u.v;
}

// B-operand (32x16 f16) from row-major B^T storage: lane holds output col
// (lane&15); K chunks {kb..kb+7} and {kb+16..kb+23}, kb = koff + (lane>>4)*8.
__device__ inline v16h ldB(const f16* p, int ld, int nbase, int koff) {
  const int lane = threadIdx.x & 31;
  const f16* q = p + (size_t)(nbase + (lane & 15)) * ld + koff + ((lane >> 4) << 3);
  F16x16 u;
  u.h[0] = *(const h8*)(q);
  u.h[1] = *(const h8*)(q + 16);
  return u.v;
}

// A-operand converted on the fly from fp32 LDS (softmax probabilities).
__device__ inline v16h ldA_f32(const float* p, int ld, int mbase, int koff) {
  const int lane = threadIdx.x & 31;
  const float* q = p + (size_t)(mbase + (lane & 15)) * ld + koff + ((lane >> 4) << 3);
  v16h v;
#pragma unroll
  for (int j = 0; j < 8; ++j) { v[j] = (f16)q[j]; v[j + 8] = (f16)q[j + 16]; }
  return v;
}

__device__ inline v8f wmma16(v16h a, v16h b, v8f c) {
  return __builtin_amdgcn_wmma_f32_16x16x32_f16(false, a, false, b, (short)0, c, false, false);
}

__device__ inline v8f vzero8() {
  v8f a;
#pragma unroll
  for (int r = 0; r < 8; ++r) a[r] = 0.0f;
  return a;
}

// ---------------- kernel 0: weight prep (f16 transpose + pad) ----------------
__global__ void prep_kernel(const float* __restrict__ wqks, const float* __restrict__ wqkq,
                            const float* __restrict__ wv,   const float* __restrict__ wout,
                            const float* __restrict__ w1,   const float* __restrict__ w2,
                            f16* __restrict__ wqks_t, f16* __restrict__ wqkq_t,
                            f16* __restrict__ wv_t,   f16* __restrict__ wout_t,
                            f16* __restrict__ w1t,    f16* __restrict__ w2t) {
  int tid = blockIdx.x * blockDim.x + threadIdx.x;
  int stride = gridDim.x * blockDim.x;
  for (int i = tid; i < 512 * 256; i += stride) {
    int c = i >> 8, k = i & 255;
    wqks_t[i] = (f16)wqks[k * 512 + c];
    wqkq_t[i] = (f16)wqkq[k * 512 + c];
  }
  for (int i = tid; i < 256 * 256; i += stride) {
    int c = i >> 8, k = i & 255;
    wv_t[i]   = (f16)wv[k * 256 + c];
    wout_t[i] = (f16)wout[k * 256 + c];
  }
  for (int i = tid; i < NE_ * 64 * 128; i += stride) {      // w1t[e][k(64)][j(128)]
    int e = i / (64 * 128), rem = i % (64 * 128);
    int k = rem >> 7, j = rem & 127;
    w1t[i] = (k < HID_ && j < NS_) ? (f16)w1[((size_t)e * NS_ + j) * HID_ + k] : (f16)0.0f;
  }
  for (int i = tid; i < NE_ * 112 * 64; i += stride) {      // w2t[e][j(112)][k(64)]
    int e = i / (112 * 64), rem = i % (112 * 64);
    int j = rem >> 6, k = rem & 63;
    w2t[i] = (j < NS_ && k < HID_) ? (f16)w2[((size_t)e * HID_ + k) * NS_ + j] : (f16)0.0f;
  }
}

// ---------------- kernel 1: QKV projections ----------------
// grid (B, 22): ty 0..6 -> q/k (w_qk_s, rows 16*ty, store i<100)
//               ty 7..10 -> q/k (w_qk_q, rows 100+16*(ty-7))
//               ty 11..21 -> v (rows 16*(ty-11)), stored transposed
__global__ void __launch_bounds__(256, 2)
proj_kernel(const float* __restrict__ x,
            const f16* __restrict__ wqks_t, const f16* __restrict__ wqkq_t,
            const f16* __restrict__ wv_t,
            f16* __restrict__ qh, f16* __restrict__ kh, f16* __restrict__ vt) {
  __shared__ f16 s_a[16 * 256];
  const int b = blockIdx.x, ty = blockIdx.y;
  int rowbase, mode;
  const f16* wt;
  if (ty < 7)       { mode = 0; rowbase = ty * 16;              wt = wqks_t; }
  else if (ty < 11) { mode = 1; rowbase = 100 + (ty - 7) * 16;  wt = wqkq_t; }
  else              { mode = 2; rowbase = (ty - 11) * 16;       wt = wv_t;   }

  for (int idx = threadIdx.x; idx < 16 * 256; idx += 256) {
    int r = idx >> 8, c = idx & 255, gr = rowbase + r;
    s_a[idx] = (gr < NTOK) ? (f16)x[((size_t)b * NTOK + gr) * 256 + c] : (f16)0.0f;
  }
  __syncthreads();

  const int wave = threadIdx.x >> 5, lane = threadIdx.x & 31;
  const int ntiles = (mode == 2) ? 16 : 32;

  // A fragments for this 16-row tile (reused for every output tile).
  v16h af[8];
#pragma unroll
  for (int kk = 0; kk < 8; ++kk) af[kk] = ldA(s_a, 256, 0, kk * 32);

  for (int t = wave; t < ntiles; t += 8) {
    // prefetch next tile's weight rows into L2 (gfx1250 global_prefetch_b8)
    if (t + 8 < ntiles)
      __builtin_prefetch(wt + (size_t)((t + 8) * 16 + (lane & 15)) * 256, 0, 1);

    // software-pipelined B fragments: 16 independent b128 loads, then chain
    v16h bf[8];
#pragma unroll
    for (int kk = 0; kk < 8; ++kk) bf[kk] = ldB(wt, 256, t * 16, kk * 32);

    v8f acc = wmma16(af[0], bf[0], vzero8());
#pragma unroll
    for (int kk = 1; kk < 8; ++kk) acc = wmma16(af[kk], bf[kk], acc);

    const int cbase = t * 16 + (lane & 15);
    const int ioff = (lane >> 4) << 3;
    if (mode == 2) {
      int h = cbase >> 5, d = cbase & 31;
      h8 pk;
#pragma unroll
      for (int r = 0; r < 8; ++r) pk[r] = (f16)acc[r];
      *(h8*)(vt + (((size_t)(b * 8 + h)) * 32 + d) * MPAD + rowbase + ioff) = pk;
    } else {
      const int c2 = (cbase < 256) ? cbase : cbase - 256;
      const int h = c2 >> 5, d = c2 & 31;
      f16* dstbase = (cbase < 256) ? qh : kh;
      f16* dst = dstbase + (((size_t)(b * 8 + h)) * NTOK) * 32 + d;
      const bool allvalid = (mode == 1) || (rowbase + 15 < NS_);
      if (allvalid) {
#pragma unroll
        for (int r = 0; r < 8; ++r) dst[(size_t)(rowbase + r + ioff) * 32] = (f16)acc[r];
      } else {
#pragma unroll
        for (int r = 0; r < 8; ++r) {
          int i = rowbase + r + ioff;
          if (i < NS_) dst[(size_t)i * 32] = (f16)acc[r];
        }
      }
    }
  }
}

// ---------------- kernel 2: fused attention + MoE per (b,h) ----------------
#define SMEM_BYTES 219200
__global__ void __launch_bounds__(256, 1)
attn_moe_kernel(const float* __restrict__ x, const unsigned char* __restrict__ mask,
                const float* __restrict__ w_gate, const float* __restrict__ b_gate,
                const float* __restrict__ ln_gamma, const float* __restrict__ ln_beta,
                const float* __restrict__ b1, const float* __restrict__ b2,
                const f16* __restrict__ qh, const f16* __restrict__ kh,
                const f16* __restrict__ vt,
                const f16* __restrict__ w1t, const f16* __restrict__ w2t,
                f16* __restrict__ ao) {
  extern __shared__ char smem[];
  float* s_dots  = (float*)smem;                 // 176 x 196 fp32
  f16*   s_q     = (f16*)(smem + 137984);        // 176 x 32
  f16*   s_k     = (f16*)(smem + 149248);        // 192 x 32
  f16*   s_vt    = (f16*)(smem + 161536);        // 32 x 192
  f16*   s_aq    = (f16*)(smem + 173824);        // 112 x 128 (relu attn_q, f16)
  f16*   s_h1    = (f16*)(smem + 202496);        // 112 x 64
  float* s_gates = (float*)(smem + 216832);      // 100 x 4
  float* s_maskf = (float*)(smem + 218432);      // 192

  const int bh = blockIdx.x, b = bh >> 3, h = bh & 7;
  const int tid = threadIdx.x, wave = tid >> 5, lane = tid & 31;

  // prefetch this block's MoE weights into L2 while staging
  if (tid < 128) {
    __builtin_prefetch(w1t + (size_t)tid * 256, 0, 1);   // covers NE*64*128
    if (tid < 112) __builtin_prefetch(w2t + (size_t)tid * 256, 0, 1);
  }

  // ---- stage Q, K, Vt, mask; compute gates ----
  for (int idx = tid; idx < MPAD * 32; idx += 256) {
    int r = idx >> 5;
    s_q[idx] = (r < NTOK) ? qh[(size_t)bh * NTOK * 32 + (size_t)r * 32 + (idx & 31)] : (f16)0.0f;
  }
  for (int idx = tid; idx < NPAD * 32; idx += 256) {
    int r = idx >> 5;
    s_k[idx] = (r < NTOK) ? kh[(size_t)bh * NTOK * 32 + (size_t)r * 32 + (idx & 31)] : (f16)0.0f;
  }
  for (int idx = tid; idx < 32 * NPAD; idx += 256) {
    int d = idx / NPAD, j = idx % NPAD;
    s_vt[idx] = (j < MPAD) ? vt[(size_t)bh * 32 * MPAD + (size_t)d * MPAD + j] : (f16)0.0f;
  }
  for (int j = tid; j < NPAD; j += 256)
    s_maskf[j] = (j < NTOK && mask[(size_t)b * NTOK + j] == 0) ? 0.0f : 1.0f;

  if (tid < NS_) {
    const int i = tid;
    float g[NE_];
#pragma unroll
    for (int e = 0; e < NE_; ++e) g[e] = b_gate[e];
    const float* xb = x + (size_t)b * (NTOK * DIM_) + h * (NTOK * D_) + i * D_;
#pragma unroll 4
    for (int d = 0; d < D_; ++d) {
      float xv = xb[d];
#pragma unroll
      for (int e = 0; e < NE_; ++e) g[e] += xv * w_gate[d * NE_ + e];
    }
    float mu = 0.25f * (g[0] + g[1] + g[2] + g[3]);
    float var = 0.0f;
#pragma unroll
    for (int e = 0; e < NE_; ++e) { float dd = g[e] - mu; var += dd * dd; }
    var *= 0.25f;
    float inv = rsqrtf(var + 1e-5f);
#pragma unroll
    for (int e = 0; e < NE_; ++e) g[e] = (g[e] - mu) * inv * ln_gamma[e] + ln_beta[e];
    float m = fmaxf(fmaxf(g[0], g[1]), fmaxf(g[2], g[3]));
    float s = 0.0f;
#pragma unroll
    for (int e = 0; e < NE_; ++e) { g[e] = __expf(g[e] - m); s += g[e]; }
    float is = 1.0f / s;
#pragma unroll
    for (int e = 0; e < NE_; ++e) s_gates[i * NE_ + e] = g[e] * is;
  }
  __syncthreads();

  // ---- dots = (Q K^T) * scale, masked -> s_dots ----
  for (int t = wave; t < 11 * 12; t += 8) {
    const int tm = t / 12, tn = t % 12;
    v8f acc = wmma16(ldA(s_q, 32, tm * 16, 0), ldB(s_k, 32, tn * 16, 0), vzero8());
    const int j = tn * 16 + (lane & 15);
    const float mf = s_maskf[j];
    const int ioff = (lane >> 4) << 3;
#pragma unroll
    for (int r = 0; r < 8; ++r) {
      int i = tm * 16 + r + ioff;
      s_dots[(size_t)i * LDSD + j] = (mf != 0.0f) ? NEG_MAX : acc[r] * 0.0625f;
    }
  }
  __syncthreads();

  // ---- stage relu(attn_q) as f16, zero padded to 112x128 ----
  for (int idx = tid; idx < 112 * 128; idx += 256) {
    int i = idx >> 7, j = idx & 127;
    float v = (i < NS_ && j < NS_) ? fmaxf(s_dots[(size_t)i * LDSD + j], 0.0f) : 0.0f;
    s_aq[idx] = (f16)v;
  }
  __syncthreads();

  // ---- expert loop ----
  for (int e = 0; e < NE_; ++e) {
    const f16* w1e = w1t + (size_t)e * 64 * 128;
    const f16* w2e = w2t + (size_t)e * 112 * 64;
    // GEMM1: h1 = relu(aq @ w1 + b1) : 112x64, K=128
    for (int t = wave; t < 7 * 4; t += 8) {
      const int tm = t >> 2, tn = t & 3;
      v16h bf[4];
#pragma unroll
      for (int kk = 0; kk < 4; ++kk) bf[kk] = ldB(w1e, 128, tn * 16, kk * 32);
      v8f acc = vzero8();
#pragma unroll
      for (int kk = 0; kk < 4; ++kk)
        acc = wmma16(ldA(s_aq, 128, tm * 16, kk * 32), bf[kk], acc);
      const int k = tn * 16 + (lane & 15);
      const float bias = (k < HID_) ? b1[e * HID_ + k] : 0.0f;
      const int ioff = (lane >> 4) << 3;
#pragma unroll
      for (int r = 0; r < 8; ++r) {
        int i = tm * 16 + r + ioff;
        float v = (k < HID_) ? fmaxf(acc[r] + bias, 0.0f) : 0.0f;
        s_h1[i * 64 + k] = (f16)v;
      }
    }
    __syncthreads();
    // GEMM2: eo = relu(h1 @ w2 + b2); dots[:NS,:NS] += gate * eo
    for (int t = wave; t < 7 * 7; t += 8) {
      const int tm = t / 7, tn = t % 7;
      v16h bf[2];
#pragma unroll
      for (int kk = 0; kk < 2; ++kk) bf[kk] = ldB(w2e, 64, tn * 16, kk * 32);
      v8f acc = vzero8();
#pragma unroll
      for (int kk = 0; kk < 2; ++kk)
        acc = wmma16(ldA(s_h1, 64, tm * 16, kk * 32), bf[kk], acc);
      const int j = tn * 16 + (lane & 15);
      const float bias = (j < NS_) ? b2[e * NS_ + j] : 0.0f;
      const int ioff = (lane >> 4) << 3;
#pragma unroll
      for (int r = 0; r < 8; ++r) {
        int i = tm * 16 + r + ioff;
        if (i < NS_ && j < NS_) {
          float v = fmaxf(acc[r] + bias, 0.0f);
          s_dots[(size_t)i * LDSD + j] += s_gates[i * NE_ + e] * v;
        }
      }
    }
    __syncthreads();
  }

  // ---- row softmax over 192 cols (masked/pad cols are -FLT_MAX -> 0) ----
  for (int row = wave; row < MPAD; row += 8) {
    float vloc[6];
    float m = NEG_MAX;
#pragma unroll
    for (int t = 0; t < 6; ++t) {
      vloc[t] = s_dots[(size_t)row * LDSD + lane + 32 * t];
      m = fmaxf(m, vloc[t]);
    }
#pragma unroll
    for (int off = 16; off > 0; off >>= 1) m = fmaxf(m, __shfl_xor(m, off));
    float s = 0.0f;
#pragma unroll
    for (int t = 0; t < 6; ++t) { vloc[t] = __expf(vloc[t] - m); s += vloc[t]; }
#pragma unroll
    for (int off = 16; off > 0; off >>= 1) s += __shfl_xor(s, off);
    float is = 1.0f / s;
#pragma unroll
    for (int t = 0; t < 6; ++t) s_dots[(size_t)row * LDSD + lane + 32 * t] = vloc[t] * is;
  }
  __syncthreads();

  // ---- out = P @ V : 176x32, K=192 ----
  for (int t = wave; t < 11 * 2; t += 8) {
    const int tm = t >> 1, tn = t & 1;
    v8f acc = vzero8();
#pragma unroll
    for (int kk = 0; kk < 6; ++kk)
      acc = wmma16(ldA_f32(s_dots, LDSD, tm * 16, kk * 32), ldB(s_vt, NPAD, tn * 16, kk * 32), acc);
    const int d = tn * 16 + (lane & 15);
    const int ioff = (lane >> 4) << 3;
#pragma unroll
    for (int r = 0; r < 8; ++r) {
      int i = tm * 16 + r + ioff;
      if (i < NTOK)
        ao[((size_t)b * NTOK + i) * DIM_ + h * 32 + d] = (f16)acc[r];
    }
  }
}

// ---------------- kernel 3: output projection ----------------
__global__ void __launch_bounds__(256, 2)
out_proj_kernel(const f16* __restrict__ ao, const f16* __restrict__ wout_t,
                const float* __restrict__ b_out, float* __restrict__ out) {
  const int rt = blockIdx.x;                 // 2624 row tiles of 16 (exact)
  const int wave = threadIdx.x >> 5, lane = threadIdx.x & 31;

  // A fragments for this block's 16 rows (shared by all 16 col tiles).
  v16h af[8];
#pragma unroll
  for (int kk = 0; kk < 8; ++kk) af[kk] = ldA(ao, 256, rt * 16, kk * 32);

  for (int t = wave; t < 16; t += 8) {
    v16h bf[8];
#pragma unroll
    for (int kk = 0; kk < 8; ++kk) bf[kk] = ldB(wout_t, 256, t * 16, kk * 32);
    v8f acc = wmma16(af[0], bf[0], vzero8());
#pragma unroll
    for (int kk = 1; kk < 8; ++kk) acc = wmma16(af[kk], bf[kk], acc);

    const int c = t * 16 + (lane & 15);
    const float bias = b_out[c];
    const int ioff = (lane >> 4) << 3;
#pragma unroll
    for (int r = 0; r < 8; ++r) {
      int row = rt * 16 + r + ioff;
      out[(size_t)row * 256 + c] = acc[r] + bias;
    }
  }
}

extern "C" void kernel_launch(void* const* d_in, const int* in_sizes, int n_in,
                              void* d_out, int out_size, void* d_ws, size_t ws_size,
                              hipStream_t stream) {
  (void)in_sizes; (void)n_in; (void)out_size; (void)ws_size;
  const float* x        = (const float*)d_in[0];
  const unsigned char* mask = (const unsigned char*)d_in[1];
  const float* w_qk_s   = (const float*)d_in[2];
  const float* w_qk_q   = (const float*)d_in[3];
  const float* w_v      = (const float*)d_in[4];
  const float* w_gate   = (const float*)d_in[5];
  const float* b_gate   = (const float*)d_in[6];
  const float* ln_gamma = (const float*)d_in[7];
  const float* ln_beta  = (const float*)d_in[8];
  const float* w1       = (const float*)d_in[9];
  const float* b1       = (const float*)d_in[10];
  const float* w2       = (const float*)d_in[11];
  const float* b2       = (const float*)d_in[12];
  const float* w_out    = (const float*)d_in[13];
  const float* b_out    = (const float*)d_in[14];
  float* out = (float*)d_out;

  char* ws = (char*)d_ws;
  f16* wqks_t = (f16*)(ws + OFF_WQKS);
  f16* wqkq_t = (f16*)(ws + OFF_WQKQ);
  f16* wv_t   = (f16*)(ws + OFF_WV);
  f16* wout_t = (f16*)(ws + OFF_WOUT);
  f16* w1t    = (f16*)(ws + OFF_W1T);
  f16* w2t    = (f16*)(ws + OFF_W2T);
  f16* qh     = (f16*)(ws + OFF_QH);
  f16* kh     = (f16*)(ws + OFF_KH);
  f16* vt     = (f16*)(ws + OFF_VT);
  f16* ao     = (f16*)(ws + OFF_AO);

  hipFuncSetAttribute((const void*)attn_moe_kernel,
                      hipFuncAttributeMaxDynamicSharedMemorySize, SMEM_BYTES);

  prep_kernel<<<256, 256, 0, stream>>>(w_qk_s, w_qk_q, w_v, w_out, w1, w2,
                                       wqks_t, wqkq_t, wv_t, wout_t, w1t, w2t);
  proj_kernel<<<dim3(B_, 22), 256, 0, stream>>>(x, wqks_t, wqkq_t, wv_t, qh, kh, vt);
  attn_moe_kernel<<<B_ * H_, 256, SMEM_BYTES, stream>>>(
      x, mask, w_gate, b_gate, ln_gamma, ln_beta, b1, b2,
      qh, kh, vt, w1t, w2t, ao);
  out_proj_kernel<<<(B_ * NTOK) / 16, 256, 0, stream>>>(ao, wout_t, b_out, out);
}
